// SocialTGCN_24670292148771
// MI455X (gfx1250) — matrix operand order repaired and checked by reading, not compiled, not run on hardware
//
#include <hip/hip_runtime.h>

typedef __bf16 bf16_t;
typedef bf16_t bfv16 __attribute__((ext_vector_type(16)));
typedef float  fv8   __attribute__((ext_vector_type(8)));

// ---------------- problem constants ----------------
#define BSZ   128
#define NP    6
#define TLEN  50
#define OUTT  25
#define MG    34560          // rows of all big GEMMs (= 768*45 = 128*45*6)

// epilogue flags (generic gemm)
#define EP_BIAS  1
#define EP_BN    2
#define EP_PRELU 4
// att-gemm flags
#define AF_BN   1
#define AF_TANH 2
#define AF_ADDY 4
#define AF_ADDS 8

#define WMMA_BF16(A_, B_, C_) \
    __builtin_amdgcn_wmma_f32_16x16x32_bf16(false, (A_), false, (B_), (short)0, (C_), false, false)

// =====================================================================
// f32 -> bf16 conversions
// =====================================================================
__global__ void cvt_f32_bf16(const float* __restrict__ in, bf16_t* __restrict__ out, int n) {
    int i = blockIdx.x * 256 + threadIdx.x;
    if (i < n) out[i] = (bf16_t)in[i];
}
// out[c*rows + r] = in[r*cols + c]   (transpose: used for tcn_w, B = W^T)
__global__ void cvt_f32_bf16_t(const float* __restrict__ in, bf16_t* __restrict__ out,
                               int rows, int cols) {
    int i = blockIdx.x * 256 + threadIdx.x;
    if (i < rows * cols) {
        int r = i / cols, c = i % cols;
        out[c * rows + r] = (bf16_t)in[i];
    }
}

// =====================================================================
// Front: windowed Conv1d (full kernel) producing skel input s0 (b, j, f)
// =====================================================================
__global__ void front_conv_kernel(const float* __restrict__ src, const float* __restrict__ cw,
                                  const float* __restrict__ cb,
                                  float* __restrict__ s0f, bf16_t* __restrict__ s0b) {
    int i = blockIdx.x * 256 + threadIdx.x;
    if (i >= 768 * 45 * 45) return;
    int o = i % 45, f = (i / 45) % 45, bp = i / (45 * 45);
    float acc = cb[o];
    const float* sp = src + ((size_t)bp * TLEN + f) * 45;
    #pragma unroll
    for (int k = 0; k < 5; ++k)
        for (int c = 0; c < 45; ++c)
            acc += sp[k * 45 + c] * cw[(o * 45 + c) * 5 + k];
    int idx = (bp * 45 + o) * 45 + f;
    s0f[idx] = acc;
    s0b[idx] = (bf16_t)acc;
}

// =====================================================================
// Adjacency (reproduces the reference's (bs*n*FN,3)->(bs*FN,n,3) reinterpret)
// =====================================================================
__device__ inline void mean_pos(const float* __restrict__ rp, int b, int e, float m[3]) {
    int p = e / 45, fr = e % 45;
    int bp = b * 6 + p;
    float s0 = 0.f, s1 = 0.f, s2 = 0.f;
    #pragma unroll
    for (int k = 0; k < 5; ++k) {
        const float* q = rp + ((size_t)bp * TLEN + fr + k) * 3;
        s0 += q[0]; s1 += q[1]; s2 += q[2];
    }
    m[0] = s0 * 0.2f; m[1] = s1 * 0.2f; m[2] = s2 * 0.2f;
}
__global__ void adjacency_kernel(const float* __restrict__ rp, float* __restrict__ adj) {
    int i = blockIdx.x * 256 + threadIdx.x;
    if (i >= BSZ * 45 * 36) return;
    int u = i % 6, v = (i / 6) % 6, f = (i / 36) % 45, b = i / (36 * 45);
    if (v == u) { adj[i] = 1.0f; return; }
    float mv[3], mu[3];
    mean_pos(rp, b, f * 6 + v, mv);
    mean_pos(rp, b, f * 6 + u, mu);
    float dx = mv[0] - mu[0], dy = mv[1] - mu[1], dz = mv[2] - mu[2];
    adj[i] = expf(-100.0f * sqrtf(dx * dx + dy * dy + dz * dz));
}

// =====================================================================
// Generic bf16 WMMA GEMM: C[MxN] = A[MxK] @ B[KxN] (row-major, fp32 acc)
// Block tile 64x128, BK=32; 256 threads = 8 waves in 2(M) x 4(N);
// each wave owns a 32x32 tile -> 4 v_wmma_f32_16x16x32_bf16 per k-step.
// Fast path (K%32==0, N%128==0): int4 global->LDS, no predication.
// Slow path: clamped address + select-zero (no exec-mask loops).
// =====================================================================
__global__ __launch_bounds__(256)
void gemm_bf16_wmma(const bf16_t* __restrict__ A, const bf16_t* __restrict__ B,
                    float* __restrict__ Cf, bf16_t* __restrict__ Cb,
                    const float* __restrict__ bias,
                    const float* __restrict__ bng, const float* __restrict__ bnb,
                    const float* __restrict__ alphap,
                    int M, int N, int K, int flags) {
    __shared__ bf16_t As[64][40];     // [m][k], 80B rows (16B aligned, conflict-free)
    __shared__ bf16_t Bs[32][136];    // [k][n], 272B rows
    int tid = threadIdx.x, lane = tid & 31, wave = tid >> 5;
    int wm = wave & 1, wn = wave >> 1;
    int mblk = blockIdx.y * 64, nblk = blockIdx.x * 128;
    int half = lane >> 4, l15 = lane & 15;
    bool fastA = ((K & 31) == 0) && ((M & 63) == 0);
    bool fastB = ((K & 31) == 0) && ((N & 127) == 0);

    fv8 z = {0.f,0.f,0.f,0.f,0.f,0.f,0.f,0.f};
    fv8 acc00 = z, acc01 = z, acc10 = z, acc11 = z;

    for (int k0 = 0; k0 < K; k0 += 32) {
        if (fastA) {
            int r = tid >> 2, cc = (tid & 3) * 8;
            *(int4*)&As[r][cc] = *(const int4*)(A + (size_t)(mblk + r) * K + k0 + cc);
        } else {
            for (int i = tid; i < 64 * 32; i += 256) {
                int r = i >> 5, c = i & 31;
                int gm = mblk + r, gk = k0 + c;
                int cm = gm < M ? gm : M - 1, ck = gk < K ? gk : K - 1;
                bf16_t vv = A[(size_t)cm * K + ck];
                As[r][c] = (gm < M && gk < K) ? vv : (bf16_t)0.0f;
            }
        }
        if (fastB) {
            #pragma unroll
            for (int it = 0; it < 2; ++it) {
                int i = tid + it * 256;
                int r = i >> 4, cc = (i & 15) * 8;
                *(int4*)&Bs[r][cc] = *(const int4*)(B + (size_t)(k0 + r) * N + nblk + cc);
            }
        } else {
            for (int i = tid; i < 32 * 128; i += 256) {
                int r = i >> 7, c = i & 127;
                int gk = k0 + r, gn = nblk + c;
                int ck = gk < K ? gk : K - 1, cn = gn < N ? gn : N - 1;
                bf16_t vv = B[(size_t)ck * N + cn];
                Bs[r][c] = (gk < K && gn < N) ? vv : (bf16_t)0.0f;
            }
        }
        __syncthreads();
        bfv16 af0, af1, bf0, bf1;
        #pragma unroll
        for (int v = 0; v < 8; ++v) {
            int k = (v < 4 ? 2 * v : 16 + 2 * (v - 4)) + 8 * half;   // ISA A layout
            af0[2*v]   = As[wm*32      + l15][k];
            af0[2*v+1] = As[wm*32      + l15][k + 1];
            af1[2*v]   = As[wm*32 + 16 + l15][k];
            af1[2*v+1] = As[wm*32 + 16 + l15][k + 1];
            int kb = (v < 4 ? 2 * v : 16 + 2 * (v - 4)) + 8 * half;   // ISA B layout
            bf0[2*v]   = Bs[kb    ][wn*32      + l15];
            bf0[2*v+1] = Bs[kb + 1][wn*32      + l15];
            bf1[2*v]   = Bs[kb    ][wn*32 + 16 + l15];
            bf1[2*v+1] = Bs[kb + 1][wn*32 + 16 + l15];
        }
        acc00 = WMMA_BF16(af0, bf0, acc00);
        acc01 = WMMA_BF16(af0, bf1, acc01);
        acc10 = WMMA_BF16(af1, bf0, acc10);
        acc11 = WMMA_BF16(af1, bf1, acc11);
        __syncthreads();
    }

    float al = (flags & EP_PRELU) ? alphap[0] : 0.0f;
    #pragma unroll
    for (int ni = 0; ni < 2; ++ni) {
        int col = nblk + wn * 32 + ni * 16 + l15;
        if (col >= N) continue;
        float bv = (flags & EP_BIAS) ? bias[col] : 0.0f;
        float g  = (flags & EP_BN)   ? bng[col]  : 1.0f;
        float b2 = (flags & EP_BN)   ? bnb[col]  : 0.0f;
        fv8 aA = ni ? acc01 : acc00;
        fv8 aB = ni ? acc11 : acc10;
        #pragma unroll
        for (int r = 0; r < 8; ++r) {
            int m0 = mblk + wm * 32 + r + 8 * half;
            int m1 = m0 + 16;
            float v0 = (aA[r] + bv) * g + b2;
            float v1 = (aB[r] + bv) * g + b2;
            if (flags & EP_PRELU) { v0 = v0 >= 0.f ? v0 : al * v0; v1 = v1 >= 0.f ? v1 : al * v1; }
            if (m0 < M) { if (Cf) Cf[(size_t)m0 * N + col] = v0; if (Cb) Cb[(size_t)m0 * N + col] = (bf16_t)v0; }
            if (m1 < M) { if (Cf) Cf[(size_t)m1 * N + col] = v1; if (Cb) Cb[(size_t)m1 * N + col] = (bf16_t)v1; }
        }
    }
}

// =====================================================================
// Batched skel att-GEMM: per sample b, Z = att(45x45) @ T[b](45xN)
// epilogue: +gbias[col]; BN per (m*256+col); tanh; +resY / +resS.
// 128 threads = 4 waves; each wave: one 16-wide N-tile x 3 M-tiles.
// =====================================================================
__global__ __launch_bounds__(128)
void attgemm_bf16_wmma(const bf16_t* __restrict__ att, const bf16_t* __restrict__ T,
                       const float* __restrict__ gbias,
                       const float* __restrict__ bng, const float* __restrict__ bnb,
                       const float* __restrict__ resY, const float* __restrict__ resS,
                       float* __restrict__ Yf, bf16_t* __restrict__ Yb,
                       int N, int flags) {
    __shared__ bf16_t As[48][72];   // att padded [m][k], K padded to 64
    __shared__ bf16_t Bs[64][72];   // [k][n]
    int b = blockIdx.y;
    int nc = blockIdx.x * 64;
    int tid = threadIdx.x, lane = tid & 31, wave = tid >> 5;
    int half = lane >> 4, l15 = lane & 15;

    for (int i = tid; i < 48 * 64; i += 128) {
        int r = i >> 6, k = i & 63;
        int cr = r < 45 ? r : 44, ck = k < 45 ? k : 44;
        bf16_t vv = att[cr * 45 + ck];
        As[r][k] = (r < 45 && k < 45) ? vv : (bf16_t)0.0f;
    }
    if ((N & 63) == 0) {
        #pragma unroll
        for (int it = 0; it < 4; ++it) {
            int i = tid + it * 128;
            int r = i >> 3, cc = (i & 7) * 8;
            int4 val = {0, 0, 0, 0};
            if (r < 45) val = *(const int4*)(T + ((size_t)b * 45 + r) * N + nc + cc);
            *(int4*)&Bs[r][cc] = val;
        }
    } else {
        for (int i = tid; i < 64 * 64; i += 128) {
            int r = i >> 6, c = i & 63;
            int col = nc + c;
            int cr = r < 45 ? r : 44, cn = col < N ? col : N - 1;
            bf16_t vv = T[((size_t)b * 45 + cr) * N + cn];
            Bs[r][c] = (r < 45 && col < N) ? vv : (bf16_t)0.0f;
        }
    }
    __syncthreads();

    fv8 z = {0.f,0.f,0.f,0.f,0.f,0.f,0.f,0.f};
    fv8 acc[3] = {z, z, z};
    #pragma unroll
    for (int ks = 0; ks < 2; ++ks) {
        bfv16 bfr;
        #pragma unroll
        for (int v = 0; v < 8; ++v) {
            int kb = (v < 4 ? 2 * v : 16 + 2 * (v - 4)) + 8 * half + 32 * ks;
            bfr[2*v]   = Bs[kb    ][wave * 16 + l15];
            bfr[2*v+1] = Bs[kb + 1][wave * 16 + l15];
        }
        #pragma unroll
        for (int mt = 0; mt < 3; ++mt) {
            bfv16 af;
            #pragma unroll
            for (int v = 0; v < 8; ++v) {
                int k = (v < 4 ? 2 * v : 16 + 2 * (v - 4)) + 8 * half + 32 * ks;
                af[2*v]   = As[mt * 16 + l15][k];
                af[2*v+1] = As[mt * 16 + l15][k + 1];
            }
            acc[mt] = WMMA_BF16(af, bfr, acc[mt]);
        }
    }

    int col = nc + wave * 16 + l15;
    if (col >= N) return;
    float bv = gbias[col];
    #pragma unroll
    for (int mt = 0; mt < 3; ++mt) {
        #pragma unroll
        for (int r = 0; r < 8; ++r) {
            int m = mt * 16 + r + 8 * half;
            if (m >= 45) continue;
            float v = acc[mt][r] + bv;
            if (flags & AF_BN)   v = v * bng[m * 256 + col] + bnb[m * 256 + col];
            if (flags & AF_TANH) v = tanhf(v);
            size_t row = (size_t)b * 45 + m;
            if (flags & AF_ADDY) v += resY[row * N + col];
            if (flags & AF_ADDS) v += resS[row * 45 + col];
            if (Yf) Yf[row * N + col] = v;
            if (Yb) Yb[row * N + col] = (bf16_t)v;
        }
    }
}

// =====================================================================
// Social adjacency aggregation (n=6). Output rows (b,f,v), bf16.
// =====================================================================
__global__ void social_agg1(const float* __restrict__ res, const float* __restrict__ adj,
                            bf16_t* __restrict__ out) {
    int i = blockIdx.x * 256 + threadIdx.x;
    if (i >= MG * 45) return;
    int c = i % 45, v = (i / 45) % 6, f = (i / 270) % 45, b = i / (270 * 45);
    const float* ar = adj + ((size_t)(b * 45 + f) * 6 + v) * 6;
    float a = 0.f;
    #pragma unroll
    for (int u = 0; u < 6; ++u)
        a += ar[u] * res[(((size_t)(b * 6 + u) * 45) + f) * 45 + c];
    out[i] = (bf16_t)a;
}
__global__ void social_agg2(const bf16_t* __restrict__ X, const float* __restrict__ adj,
                            bf16_t* __restrict__ out, int C) {
    size_t i = (size_t)blockIdx.x * 256 + threadIdx.x;
    if (i >= (size_t)MG * C) return;
    int c = (int)(i % C);
    int row = (int)(i / C);
    int v = row % 6, f = (row / 6) % 45, b = row / 270;
    const float* ar = adj + ((size_t)(b * 45 + f) * 6 + v) * 6;
    float a = 0.f;
    #pragma unroll
    for (int u = 0; u < 6; ++u)
        a += ar[u] * (float)X[((size_t)(b * 45 + f) * 6 + u) * C + c];
    out[i] = (bf16_t)a;
}

// =====================================================================
// tx input: txin[b,ch,v,w] = res[b,v,w,ch] + xfinal[b,f=w,v,c=ch]
// =====================================================================
__global__ void txin_kernel(const float* __restrict__ res, const bf16_t* __restrict__ xs,
                            float* __restrict__ txin) {
    int i = blockIdx.x * 256 + threadIdx.x;
    if (i >= BSZ * 45 * 6 * 45) return;
    int w = i % 45, v = (i / 45) % 6, ch = (i / 270) % 45, b = i / (270 * 45);
    float r = res[(((size_t)(b * 6 + v) * 45) + w) * 45 + ch];
    float x = (float)xs[(((size_t)(b * 45 + w) * 6) + v) * 45 + ch];
    txin[i] = r + x;
}

// =====================================================================
// tx 3x3 conv over (H=6, W=45), SAME pad, BN+PReLU, optional residual.
// =====================================================================
__global__ void tx_conv_kernel(const float* __restrict__ in, int Cin,
                               const float* __restrict__ w, const float* __restrict__ bias,
                               const float* __restrict__ bng, const float* __restrict__ bnb,
                               const float* __restrict__ alphap,
                               const float* __restrict__ resadd,
                               float* __restrict__ out, float* __restrict__ dout) {
    int i = blockIdx.x * 256 + threadIdx.x;
    if (i >= BSZ * OUTT * 6 * 45) return;
    int x = i % 45, v = (i / 45) % 6, t = (i / 270) % OUTT, b = i / (270 * OUTT);
    float acc = bias[t];
    for (int ci = 0; ci < Cin; ++ci) {
        const float* ip = in + ((size_t)(b * Cin + ci) * 6) * 45;
        const float* wp = w + ((size_t)(t * Cin + ci)) * 9;
        #pragma unroll
        for (int kh = -1; kh <= 1; ++kh) {
            int vv = v + kh;
            if (vv < 0 || vv >= 6) continue;
            #pragma unroll
            for (int kw = -1; kw <= 1; ++kw) {
                int ww = x + kw;
                if (ww < 0 || ww >= 45) continue;
                acc += ip[vv * 45 + ww] * wp[(kh + 1) * 3 + (kw + 1)];
            }
        }
    }
    acc = acc * bng[t] + bnb[t];
    float al = alphap[0];
    acc = acc >= 0.f ? acc : al * acc;
    if (resadd) acc += resadd[i];
    if (dout) dout[(((size_t)(b * 6 + v) * OUTT) + t) * 45 + x] = acc;
    else      out[i] = acc;
}

// =====================================================================
// host side
// =====================================================================
enum {
    IN_SRC = 0, IN_ROOT, IN_CONV_W, IN_CONV_B,
    IN_GIN_ATT, IN_GIN_W, IN_GIN_B, IN_BNIN_G, IN_BNIN_B,
    IN_B0_G1_ATT, IN_B0_G1_W, IN_B0_G1_B, IN_B0_BN1_G, IN_B0_BN1_B,
    IN_B0_G2_ATT, IN_B0_G2_W, IN_B0_G2_B, IN_B0_BN2_G, IN_B0_BN2_B,
    IN_B1_G1_ATT, IN_B1_G1_W, IN_B1_G1_B, IN_B1_BN1_G, IN_B1_BN1_B,
    IN_B1_G2_ATT, IN_B1_G2_W, IN_B1_G2_B, IN_B1_BN2_G, IN_B1_BN2_B,
    IN_GOUT_ATT, IN_GOUT_W, IN_GOUT_B,
    IN_S0_ENCW, IN_S0_ENCB, IN_S0_BN1G, IN_S0_BN1B, IN_S0_ALPHA, IN_S0_TCNW, IN_S0_TCNB, IN_S0_BN2G, IN_S0_BN2B,
    IN_S1_ENCW, IN_S1_ENCB, IN_S1_BN1G, IN_S1_BN1B, IN_S1_ALPHA, IN_S1_TCNW, IN_S1_TCNB, IN_S1_BN2G, IN_S1_BN2B,
    IN_S2_ENCW, IN_S2_ENCB, IN_S2_BN1G, IN_S2_BN1B, IN_S2_ALPHA, IN_S2_TCNW, IN_S2_TCNB, IN_S2_BN2G, IN_S2_BN2B,
    IN_T0_W, IN_T0_B, IN_T0_BNG, IN_T0_BNB, IN_T0_ALPHA,
    IN_T1_W, IN_T1_B, IN_T1_BNG, IN_T1_BNB, IN_T1_ALPHA,
    IN_T2_W, IN_T2_B, IN_T2_BNG, IN_T2_BNB, IN_T2_ALPHA,
    IN_T3_W, IN_T3_B, IN_T3_BNG, IN_T3_BNB, IN_T3_ALPHA
};

extern "C" void kernel_launch(void* const* d_in, const int* in_sizes, int n_in,
                              void* d_out, int out_size, void* d_ws, size_t ws_size,
                              hipStream_t stream) {
    (void)in_sizes; (void)n_in; (void)out_size; (void)ws_size;
    char* ws = (char*)d_ws;
    size_t off = 0;
    auto alloc = [&](size_t bytes) -> size_t {
        size_t o = off; off += (bytes + 255) & ~(size_t)255; return o;
    };
    const size_t O_RES = alloc(1555200 * sizeof(float));   // res_x (768,45,45)
    const size_t O_ADJ = alloc(207360 * sizeof(float));    // adjacency
    const size_t O_S0F = alloc(1555200 * sizeof(float));   // skel input fp32 -> later txin
    const size_t O_S0B = alloc(1555200 * sizeof(bf16_t));  // skel input bf16
    const size_t O_YF  = alloc(8847360 * sizeof(float));   // Yf fp32 -> later Aagg bf16 / t0,t1
    const size_t O_TB  = alloc(8847360 * sizeof(bf16_t));  // Tb -> later Henc (spans O_TB+O_YB)
    const size_t O_YB  = alloc(8847360 * sizeof(bf16_t));  // Yb (contiguous after O_TB)
    const size_t O_XS  = alloc(17694720 * sizeof(bf16_t)); // Zb (first half) -> Xs
    // bf16 weights
    const size_t W_GIN_ATT = alloc(2025 * 2), W_B0G1_ATT = alloc(2025 * 2),
                 W_B0G2_ATT = alloc(2025 * 2), W_B1G1_ATT = alloc(2025 * 2),
                 W_B1G2_ATT = alloc(2025 * 2), W_GOUT_ATT = alloc(2025 * 2);
    const size_t W_GIN_W = alloc(11520 * 2), W_B0G1_W = alloc(65536 * 2),
                 W_B0G2_W = alloc(65536 * 2), W_B1G1_W = alloc(65536 * 2),
                 W_B1G2_W = alloc(65536 * 2), W_GOUT_W = alloc(11520 * 2);
    const size_t W_ENC1 = alloc(23040 * 2), W_ENC2 = alloc(262144 * 2), W_ENC3 = alloc(23040 * 2);
    const size_t W_TCN1 = alloc(262144 * 2), W_TCN2 = alloc(262144 * 2), W_TCN3 = alloc(2025 * 2);

    auto F = [&](int i) { return (const float*)d_in[i]; };
    auto cv = [&](int idx, size_t o, int n) {
        cvt_f32_bf16<<<dim3((n + 255) / 256), dim3(256), 0, stream>>>(F(idx), (bf16_t*)(ws + o), n);
    };
    auto cvt = [&](int idx, size_t o, int rows, int cols) {
        cvt_f32_bf16_t<<<dim3((rows * cols + 255) / 256), dim3(256), 0, stream>>>(
            F(idx), (bf16_t*)(ws + o), rows, cols);
    };
    // weight conversion
    cv(IN_GIN_ATT, W_GIN_ATT, 2025);    cv(IN_B0_G1_ATT, W_B0G1_ATT, 2025);
    cv(IN_B0_G2_ATT, W_B0G2_ATT, 2025); cv(IN_B1_G1_ATT, W_B1G1_ATT, 2025);
    cv(IN_B1_G2_ATT, W_B1G2_ATT, 2025); cv(IN_GOUT_ATT, W_GOUT_ATT, 2025);
    cv(IN_GIN_W, W_GIN_W, 11520);       cv(IN_B0_G1_W, W_B0G1_W, 65536);
    cv(IN_B0_G2_W, W_B0G2_W, 65536);    cv(IN_B1_G1_W, W_B1G1_W, 65536);
    cv(IN_B1_G2_W, W_B1G2_W, 65536);    cv(IN_GOUT_W, W_GOUT_W, 11520);
    cv(IN_S0_ENCW, W_ENC1, 23040);      cv(IN_S1_ENCW, W_ENC2, 262144);
    cv(IN_S2_ENCW, W_ENC3, 23040);
    cvt(IN_S0_TCNW, W_TCN1, 512, 512);  cvt(IN_S1_TCNW, W_TCN2, 512, 512);
    cvt(IN_S2_TCNW, W_TCN3, 45, 45);

    // front conv + adjacency
    front_conv_kernel<<<dim3(6075), dim3(256), 0, stream>>>(
        F(IN_SRC), F(IN_CONV_W), F(IN_CONV_B), (float*)(ws + O_S0F), (bf16_t*)(ws + O_S0B));
    adjacency_kernel<<<dim3(810), dim3(256), 0, stream>>>(F(IN_ROOT), (float*)(ws + O_ADJ));

    float*  Yf   = (float*)(ws + O_YF);
    bf16_t* Tb   = (bf16_t*)(ws + O_TB);
    bf16_t* Yb   = (bf16_t*)(ws + O_YB);
    bf16_t* Zb   = (bf16_t*)(ws + O_XS);
    float*  resf = (float*)(ws + O_RES);
    const float* adjp = (const float*)(ws + O_ADJ);

    auto gemm = [&](size_t aoff, size_t boff, float* cf, bf16_t* cb,
                    const float* bias, const float* g, const float* b2, const float* al,
                    int M, int N, int K, int fl) {
        gemm_bf16_wmma<<<dim3((N + 127) / 128, (M + 63) / 64), dim3(256), 0, stream>>>(
            (const bf16_t*)(ws + aoff), (const bf16_t*)(ws + boff), cf, cb, bias, g, b2, al,
            M, N, K, fl);
    };
    auto attg = [&](size_t attoff, const float* gb, const float* g, const float* b2,
                    const float* rY, const float* rS, float* yf, bf16_t* yb, int N, int fl) {
        attgemm_bf16_wmma<<<dim3((N + 63) / 64, 768), dim3(128), 0, stream>>>(
            (const bf16_t*)(ws + attoff), Tb, gb, g, b2, rY, rS, yf, yb, N, fl);
    };

    // ---- skel GCN ----
    gemm(O_S0B, W_GIN_W, nullptr, Tb, nullptr, nullptr, nullptr, nullptr, MG, 256, 45, 0);
    attg(W_GIN_ATT, F(IN_GIN_B), F(IN_BNIN_G), F(IN_BNIN_B), nullptr, nullptr,
         Yf, Yb, 256, AF_BN | AF_TANH);
    // block 0
    gemm(O_YB, W_B0G1_W, nullptr, Tb, nullptr, nullptr, nullptr, nullptr, MG, 256, 256, 0);
    attg(W_B0G1_ATT, F(IN_B0_G1_B), F(IN_B0_BN1_G), F(IN_B0_BN1_B), nullptr, nullptr,
         nullptr, Zb, 256, AF_BN | AF_TANH);
    gemm(O_XS, W_B0G2_W, nullptr, Tb, nullptr, nullptr, nullptr, nullptr, MG, 256, 256, 0);
    attg(W_B0G2_ATT, F(IN_B0_G2_B), F(IN_B0_BN2_G), F(IN_B0_BN2_B), Yf, nullptr,
         Yf, Yb, 256, AF_BN | AF_TANH | AF_ADDY);
    // block 1
    gemm(O_YB, W_B1G1_W, nullptr, Tb, nullptr, nullptr, nullptr, nullptr, MG, 256, 256, 0);
    attg(W_B1G1_ATT, F(IN_B1_G1_B), F(IN_B1_BN1_G), F(IN_B1_BN1_B), nullptr, nullptr,
         nullptr, Zb, 256, AF_BN | AF_TANH);
    gemm(O_XS, W_B1G2_W, nullptr, Tb, nullptr, nullptr, nullptr, nullptr, MG, 256, 256, 0);
    attg(W_B1G2_ATT, F(IN_B1_G2_B), F(IN_B1_BN2_G), F(IN_B1_BN2_B), Yf, nullptr,
         Yf, Yb, 256, AF_BN | AF_TANH | AF_ADDY);
    // gc_out + residual -> res_x
    gemm(O_YB, W_GOUT_W, nullptr, Tb, nullptr, nullptr, nullptr, nullptr, MG, 45, 256, 0);
    attg(W_GOUT_ATT, F(IN_GOUT_B), nullptr, nullptr, nullptr, (const float*)(ws + O_S0F),
         resf, nullptr, 45, AF_ADDS);

    // ---- social stack (buffers reuse freed skel regions) ----
    bf16_t* Aagg = (bf16_t*)(ws + O_YF);   // 34560 x 512 bf16
    bf16_t* Henc = (bf16_t*)(ws + O_TB);   // 34560 x 512 bf16 (spans O_TB+O_YB)
    bf16_t* Xs   = (bf16_t*)(ws + O_XS);   // 34560 x 512 bf16

    social_agg1<<<dim3(6075), dim3(256), 0, stream>>>(resf, adjp, Aagg);
    gemm(O_YF, W_ENC1, nullptr, Henc, F(IN_S0_ENCB), F(IN_S0_BN1G), F(IN_S0_BN1B),
         F(IN_S0_ALPHA), MG, 512, 45, EP_BIAS | EP_BN | EP_PRELU);
    gemm(O_TB, W_TCN1, nullptr, Xs, F(IN_S0_TCNB), F(IN_S0_BN2G), F(IN_S0_BN2B),
         nullptr, MG, 512, 512, EP_BIAS | EP_BN);

    social_agg2<<<dim3(69120), dim3(256), 0, stream>>>(Xs, adjp, Aagg, 512);
    gemm(O_YF, W_ENC2, nullptr, Henc, F(IN_S1_ENCB), F(IN_S1_BN1G), F(IN_S1_BN1B),
         F(IN_S1_ALPHA), MG, 512, 512, EP_BIAS | EP_BN | EP_PRELU);
    gemm(O_TB, W_TCN2, nullptr, Xs, F(IN_S1_TCNB), F(IN_S1_BN2G), F(IN_S1_BN2B),
         nullptr, MG, 512, 512, EP_BIAS | EP_BN);

    social_agg2<<<dim3(69120), dim3(256), 0, stream>>>(Xs, adjp, Aagg, 512);
    gemm(O_YF, W_ENC3, nullptr, Henc, F(IN_S2_ENCB), F(IN_S2_BN1G), F(IN_S2_BN1B),
         F(IN_S2_ALPHA), MG, 45, 512, EP_BIAS | EP_BN | EP_PRELU);
    gemm(O_TB, W_TCN3, nullptr, Xs, F(IN_S2_TCNB), F(IN_S2_BN2G), F(IN_S2_BN2B),
         nullptr, MG, 45, 45, EP_BIAS | EP_BN);

    // ---- tx conv stack ----
    float* txin = (float*)(ws + O_S0F);
    float* t0 = (float*)(ws + O_YF);
    float* t1 = (float*)(ws + O_YF + 3456000);
    txin_kernel<<<dim3(6075), dim3(256), 0, stream>>>(resf, Xs, txin);
    tx_conv_kernel<<<dim3(3375), dim3(256), 0, stream>>>(
        txin, 45, F(IN_T0_W), F(IN_T0_B), F(IN_T0_BNG), F(IN_T0_BNB), F(IN_T0_ALPHA),
        nullptr, t0, nullptr);
    tx_conv_kernel<<<dim3(3375), dim3(256), 0, stream>>>(
        t0, 25, F(IN_T1_W), F(IN_T1_B), F(IN_T1_BNG), F(IN_T1_BNB), F(IN_T1_ALPHA),
        t0, t1, nullptr);
    tx_conv_kernel<<<dim3(3375), dim3(256), 0, stream>>>(
        t1, 25, F(IN_T2_W), F(IN_T2_B), F(IN_T2_BNG), F(IN_T2_BNB), F(IN_T2_ALPHA),
        t1, t0, nullptr);
    tx_conv_kernel<<<dim3(3375), dim3(256), 0, stream>>>(
        t0, 25, F(IN_T3_W), F(IN_T3_B), F(IN_T3_BNG), F(IN_T3_BNB), F(IN_T3_ALPHA),
        t0, t1, (float*)d_out);
}
